// MemoryTransformerLayer_21182778704144
// MI455X (gfx1250) — matrix operand order repaired and checked by reading, not compiled, not run on hardware
//
#include <hip/hip_runtime.h>

// ---------------------------------------------------------------------------
// Types / helpers
// ---------------------------------------------------------------------------
typedef __attribute__((ext_vector_type(16))) __bf16 bf16x16;
typedef __attribute__((ext_vector_type(8)))  float  f32x8;
typedef unsigned short bfu;   // bf16 storage bits

#define WMMA_BF16(A_, B_, C_) \
  __builtin_amdgcn_wmma_f32_16x16x32_bf16(false, (A_), false, (B_), (short)0, (C_), false, false)

union FragU {
  bf16x16 v;
  uint4   q[2];
  bfu     s[16];
};

__device__ __forceinline__ bfu f2bf(float f) {
  unsigned u = __float_as_uint(f);
  u += 0x7FFFu + ((u >> 16) & 1u);        // round-to-nearest-even
  return (bfu)(u >> 16);
}
__device__ __forceinline__ float bf2f(bfu b) {
  return __uint_as_float(((unsigned)b) << 16);
}

// A fragment (16x32 bf16, M x K), row-major source, base points at (row0,k0).
// lane<16: row=lane, K = {0..7, 16..23}; lane>=16: row=lane-16, K = {8..15, 24..31}
__device__ __forceinline__ bf16x16 load_a_frag(const bfu* base, int ld) {
  int lane = threadIdx.x & 31;
  int m  = lane & 15;
  int kb = (lane < 16) ? 0 : 8;
  const bfu* p = base + (size_t)m * ld + kb;
  FragU f;
  f.q[0] = *(const uint4*)(p);
  f.q[1] = *(const uint4*)(p + 16);
  return f.v;
}

// B fragment (32x16 bf16, K x N) for C = A * W^T where W is [N,K] row-major:
// B[k][n] = W[n0+n][k0+k].  lane<16: K={0..7,16..23} ; lane>=16: K={8..15,24..31} ; n = lane&15
__device__ __forceinline__ bf16x16 load_b_frag_wt(const bfu* W, int ldw, int n0, int k0) {
  int lane = threadIdx.x & 31;
  int n  = lane & 15;
  int kb = (lane < 16) ? 0 : 8;
  const bfu* p = W + (size_t)(n0 + n) * ldw + k0 + kb;
  FragU f;
  f.q[0] = *(const uint4*)(p);
  f.q[1] = *(const uint4*)(p + 16);
  return f.v;
}

// A fragment from an LDS float tile [16][32] (probabilities), converted to bf16
__device__ __forceinline__ bf16x16 load_a_frag_lds(const float* p) {
  int lane = threadIdx.x & 31;
  int m  = lane & 15;
  int kb = (lane < 16) ? 0 : 8;
  FragU f;
#pragma unroll
  for (int e = 0; e < 16; ++e) {
    int k = kb + e + ((e < 8) ? 0 : 8);
    f.s[e] = f2bf(p[m * 32 + k]);
  }
  return f.v;
}

// ---------------------------------------------------------------------------
// K0: fp32 -> bf16 converters
// ---------------------------------------------------------------------------
__global__ void mtl_cvt_bf16(const float* __restrict__ src, bfu* __restrict__ dst, int n) {
  int i = blockIdx.x * blockDim.x + threadIdx.x;
  if (i < n) dst[i] = f2bf(src[i]);
}

// kvb = concat(mems_x, x) per batch, bf16  -> [B][1536][512]
__global__ void mtl_build_kvb(const float* __restrict__ mems_x, const float* __restrict__ x,
                              bfu* __restrict__ kvb) {
  size_t i = (size_t)blockIdx.x * blockDim.x + threadIdx.x;   // over 8*1536*512
  int col = (int)(i & 511);
  size_t row = i >> 9;
  int b = (int)(row / 1536);
  int r = (int)(row % 1536);
  float v = (r < 512) ? mems_x[((size_t)b * 512 + r) * 512 + col]
                      : x[((size_t)b * 1024 + (r - 512)) * 512 + col];
  kvb[i] = f2bf(v);
}

// ---------------------------------------------------------------------------
// K1: QKV projection (WMMA GEMM). Each wave computes a 16x64 strip = one head.
//   q -> [B,H,S,64]   k -> [B,H,1536,64]   v -> [B,H,64,1536]  (V transposed!)
// ---------------------------------------------------------------------------
__global__ __launch_bounds__(256) void mtl_qkv_kernel(
    const bfu* __restrict__ kvb, const bfu* __restrict__ wqkv,
    const float* __restrict__ in_proj_b,
    bfu* __restrict__ qbuf, bfu* __restrict__ kbuf, bfu* __restrict__ vbuf) {
  int proj = blockIdx.y;                              // 0=q 1=k 2=v
  int job  = blockIdx.x * 8 + (threadIdx.x >> 5);     // wave-uniform
  int Mtiles = (proj == 0) ? 512 : 768;               // q: 8192 rows, kv: 12288 rows
  if (job >= Mtiles * 8) return;                      // 8 head-strips per row-tile
  int mt = job >> 3, h = job & 7;
  int n0 = h * 64;
  const bfu* W = wqkv + (size_t)proj * 512 * 512;
  const float* bias = in_proj_b + proj * 512;

  int gm = mt * 16, b, t;
  size_t arow;
  if (proj == 0) { b = gm >> 10; t = gm & 1023; arow = (size_t)b * 1536 + 512 + t; }
  else           { b = gm / 1536; t = gm % 1536; arow = (size_t)gm; }

  f32x8 c0 = {}, c1 = {}, c2 = {}, c3 = {};
  for (int k0 = 0; k0 < 512; k0 += 32) {
    bf16x16 a = load_a_frag(kvb + arow * 512 + k0, 512);
    c0 = WMMA_BF16(a, load_b_frag_wt(W, 512, n0,      k0), c0);
    c1 = WMMA_BF16(a, load_b_frag_wt(W, 512, n0 + 16, k0), c1);
    c2 = WMMA_BF16(a, load_b_frag_wt(W, 512, n0 + 32, k0), c2);
    c3 = WMMA_BF16(a, load_b_frag_wt(W, 512, n0 + 48, k0), c3);
  }
  int lane = threadIdx.x & 31;
  int n = lane & 15, mb = (lane < 16) ? 0 : 8;
  float bi0 = bias[n0 + n], bi1 = bias[n0 + 16 + n];
  float bi2 = bias[n0 + 32 + n], bi3 = bias[n0 + 48 + n];
  size_t hb = (size_t)b * 8 + h;
#pragma unroll
  for (int r = 0; r < 8; ++r) {
    int tok = t + mb + r;
    bfu v0 = f2bf(c0[r] + bi0), v1 = f2bf(c1[r] + bi1);
    bfu v2 = f2bf(c2[r] + bi2), v3 = f2bf(c3[r] + bi3);
    if (proj == 0) {
      bfu* d = qbuf + (hb * 1024 + tok) * 64;
      d[n] = v0; d[16 + n] = v1; d[32 + n] = v2; d[48 + n] = v3;
    } else if (proj == 1) {
      bfu* d = kbuf + (hb * 1536 + tok) * 64;
      d[n] = v0; d[16 + n] = v1; d[32 + n] = v2; d[48 + n] = v3;
    } else {            // V stored transposed: [B,H][dh][1536]
      bfu* d = vbuf + hb * 64 * 1536 + tok;
      d[(size_t)(n)      * 1536] = v0;
      d[(size_t)(16 + n) * 1536] = v1;
      d[(size_t)(32 + n) * 1536] = v2;
      d[(size_t)(48 + n) * 1536] = v3;
    }
  }
}

// B fragment from transposed V [dh][1536]: B[k][n] = Vt[dh0+n][key0+k] (contiguous)
__device__ __forceinline__ bf16x16 load_b_frag_vt(const bfu* vt, int key0, int dh0) {
  int lane = threadIdx.x & 31;
  int n  = lane & 15;
  int kb = (lane < 16) ? 0 : 8;
  const bfu* p = vt + (size_t)(dh0 + n) * 1536 + key0 + kb;
  FragU f;
  f.q[0] = *(const uint4*)(p);
  f.q[1] = *(const uint4*)(p + 16);
  return f.v;
}

// ---------------------------------------------------------------------------
// K2: fused attention (softmax over 1536 keys) + head-avg entropy
//   block = 256 threads = 8 waves, wave w handles head w for one 16-query tile
// ---------------------------------------------------------------------------
__global__ __launch_bounds__(256) void mtl_attn_kernel(
    const bfu* __restrict__ qb, const bfu* __restrict__ kb, const bfu* __restrict__ vb,
    bfu* __restrict__ attn_cat, float* __restrict__ ent_cur) {
  __shared__ float lds_p[8][16][32];     // per-head 16q x 32key probability tile

  const int b  = blockIdx.x >> 6;        // 64 q-tiles per batch
  const int qt = blockIdx.x & 63;
  const int w  = threadIdx.x >> 5;       // head
  const int lane = threadIdx.x & 31;

  const bfu* qh = qb + (((size_t)b * 8 + w) * 1024 + (size_t)qt * 16) * 64;
  const bfu* kh = kb + (((size_t)b * 8 + w) * 1536) * 64;
  const bfu* vh = vb + ((size_t)b * 8 + w) * 64 * 1536;   // transposed V

  bf16x16 a0 = load_a_frag(qh, 64);        // dh 0..31
  bf16x16 a1 = load_a_frag(qh + 32, 64);   // dh 32..63
  const float scale = 0.125f;              // 1/sqrt(64)

  // ---- pass A: row max ----
  float vmax[8];
#pragma unroll
  for (int r = 0; r < 8; ++r) vmax[r] = -3.0e38f;
  for (int kt = 0; kt < 96; ++kt) {
    bf16x16 b0 = load_b_frag_wt(kh, 64, kt * 16, 0);
    bf16x16 b1 = load_b_frag_wt(kh, 64, kt * 16, 32);
    f32x8 c = {};
    c = WMMA_BF16(a0, b0, c);
    c = WMMA_BF16(a1, b1, c);
#pragma unroll
    for (int r = 0; r < 8; ++r) vmax[r] = fmaxf(vmax[r], c[r] * scale);
  }
  float rowM[8];
#pragma unroll
  for (int r = 0; r < 8; ++r) {
    float v = vmax[r];
    v = fmaxf(v, __shfl_xor(v, 1, 32));
    v = fmaxf(v, __shfl_xor(v, 2, 32));
    v = fmaxf(v, __shfl_xor(v, 4, 32));
    v = fmaxf(v, __shfl_xor(v, 8, 32));
    rowM[r] = v;
  }

  // ---- pass B: sum of exp ----
  float vsum[8];
#pragma unroll
  for (int r = 0; r < 8; ++r) vsum[r] = 0.f;
  for (int kt = 0; kt < 96; ++kt) {
    bf16x16 b0 = load_b_frag_wt(kh, 64, kt * 16, 0);
    bf16x16 b1 = load_b_frag_wt(kh, 64, kt * 16, 32);
    f32x8 c = {};
    c = WMMA_BF16(a0, b0, c);
    c = WMMA_BF16(a1, b1, c);
#pragma unroll
    for (int r = 0; r < 8; ++r) vsum[r] += __expf(c[r] * scale - rowM[r]);
  }
  float invL[8];
#pragma unroll
  for (int r = 0; r < 8; ++r) {
    float v = vsum[r];
    v += __shfl_xor(v, 1, 32);
    v += __shfl_xor(v, 2, 32);
    v += __shfl_xor(v, 4, 32);
    v += __shfl_xor(v, 8, 32);
    invL[r] = 1.f / v;
  }

  // ---- pass C: normalized probs -> entropy + O = P*V ----
  f32x8 o0 = {}, o1 = {}, o2 = {}, o3 = {};
  float entacc = 0.f;
  const int em = threadIdx.x >> 4;   // 0..15: query row owned for entropy
  const int en = threadIdx.x & 15;
  const int cn = lane & 15;
  const int cm = (lane < 16) ? 0 : 8;

  for (int kc = 0; kc < 48; ++kc) {
    int key0 = kc * 32;
#pragma unroll
    for (int sub = 0; sub < 2; ++sub) {
      bf16x16 b0 = load_b_frag_wt(kh, 64, key0 + sub * 16, 0);
      bf16x16 b1 = load_b_frag_wt(kh, 64, key0 + sub * 16, 32);
      f32x8 c = {};
      c = WMMA_BF16(a0, b0, c);
      c = WMMA_BF16(a1, b1, c);
#pragma unroll
      for (int r = 0; r < 8; ++r)
        lds_p[w][cm + r][sub * 16 + cn] = __expf(c[r] * scale - rowM[r]) * invL[r];
    }
    __syncthreads();

    // head-averaged entropy contribution (each thread owns (em, en) and (em, en+16))
#pragma unroll
    for (int j = 0; j < 32; j += 16) {
      float pb = 0.f;
#pragma unroll
      for (int hh = 0; hh < 8; ++hh) pb += lds_p[hh][em][en + j];
      pb *= 0.125f;
      entacc += pb * __logf(pb + 1e-8f);
    }

    // O accumulation: A = P tile (16x32), B = V^T (32 keys x 16 dh, contiguous loads)
    bf16x16 ap = load_a_frag_lds(&lds_p[w][0][0]);
    o0 = WMMA_BF16(ap, load_b_frag_vt(vh, key0, 0),  o0);
    o1 = WMMA_BF16(ap, load_b_frag_vt(vh, key0, 16), o1);
    o2 = WMMA_BF16(ap, load_b_frag_vt(vh, key0, 32), o2);
    o3 = WMMA_BF16(ap, load_b_frag_vt(vh, key0, 48), o3);
    __syncthreads();
  }

  // store O tile (bf16, concat-head layout [B,S,512])
  {
    size_t rowbase = (size_t)b * 1024 + (size_t)qt * 16;
#pragma unroll
    for (int r = 0; r < 8; ++r) {
      bfu* dst = attn_cat + (rowbase + cm + r) * 512 + w * 64;
      dst[0  + cn] = f2bf(o0[r]);
      dst[16 + cn] = f2bf(o1[r]);
      dst[32 + cn] = f2bf(o2[r]);
      dst[48 + cn] = f2bf(o3[r]);
    }
  }

  // entropy reduce: 16 threads sharing em live in one 16-lane half
  float e = entacc;
  e += __shfl_xor(e, 1, 32);
  e += __shfl_xor(e, 2, 32);
  e += __shfl_xor(e, 4, 32);
  e += __shfl_xor(e, 8, 32);
  if ((lane & 15) == 0)
    ent_cur[(size_t)b * 1024 + qt * 16 + em] = -e;
}

// ---------------------------------------------------------------------------
// K3: generic WMMA GEMM  C[M,N] = A[M,K] * W[N,K]^T + bias
//   Each wave computes a 16x64 strip (A fragment reused across 4 N-tiles).
//   mode 0: store f32 to outF ; mode 1: relu -> bf16 to outB
// ---------------------------------------------------------------------------
__global__ __launch_bounds__(256) void mtl_gemm_wt(
    const bfu* __restrict__ A, int lda, const bfu* __restrict__ W, int ldw,
    const float* __restrict__ bias, float* __restrict__ outF, bfu* __restrict__ outB,
    int Mtiles, int Ngroups, int K, int mode) {
  int job = blockIdx.x * 8 + (threadIdx.x >> 5);
  if (job >= Mtiles * Ngroups) return;
  int mt = job / Ngroups, ng = job % Ngroups;
  int m0 = mt * 16, n0 = ng * 64;
  f32x8 c0 = {}, c1 = {}, c2 = {}, c3 = {};
  for (int k0 = 0; k0 < K; k0 += 32) {
    bf16x16 a = load_a_frag(A + (size_t)m0 * lda + k0, lda);
    c0 = WMMA_BF16(a, load_b_frag_wt(W, ldw, n0,      k0), c0);
    c1 = WMMA_BF16(a, load_b_frag_wt(W, ldw, n0 + 16, k0), c1);
    c2 = WMMA_BF16(a, load_b_frag_wt(W, ldw, n0 + 32, k0), c2);
    c3 = WMMA_BF16(a, load_b_frag_wt(W, ldw, n0 + 48, k0), c3);
  }
  int lane = threadIdx.x & 31;
  int n = lane & 15, mb = (lane < 16) ? 0 : 8;
  int N = Ngroups * 64;
  float bi0 = bias[n0 + n], bi1 = bias[n0 + 16 + n];
  float bi2 = bias[n0 + 32 + n], bi3 = bias[n0 + 48 + n];
#pragma unroll
  for (int r = 0; r < 8; ++r) {
    size_t off = (size_t)(m0 + mb + r) * N + n0 + n;
    float v0 = c0[r] + bi0, v1 = c1[r] + bi1, v2 = c2[r] + bi2, v3 = c3[r] + bi3;
    if (mode == 0) {
      outF[off] = v0; outF[off + 16] = v1; outF[off + 32] = v2; outF[off + 48] = v3;
    } else {
      outB[off]      = f2bf(fmaxf(v0, 0.f));
      outB[off + 16] = f2bf(fmaxf(v1, 0.f));
      outB[off + 32] = f2bf(fmaxf(v2, 0.f));
      outB[off + 48] = f2bf(fmaxf(v3, 0.f));
    }
  }
}

// ---------------------------------------------------------------------------
// K4: residual add + LayerNorm (one block per row of 512)
// ---------------------------------------------------------------------------
__global__ __launch_bounds__(256) void mtl_addln(
    const float* __restrict__ resid, const float* __restrict__ delta,
    const float* __restrict__ g, const float* __restrict__ be,
    float* __restrict__ outF, bfu* __restrict__ outB) {
  __shared__ float red[10];
  int tid = threadIdx.x;
  size_t base = (size_t)blockIdx.x * 512;
  float v0 = resid[base + tid]       + delta[base + tid];
  float v1 = resid[base + 256 + tid] + delta[base + 256 + tid];

  float s = v0 + v1;
  for (int m = 16; m >= 1; m >>= 1) s += __shfl_xor(s, m, 32);
  if ((tid & 31) == 0) red[tid >> 5] = s;
  __syncthreads();
  if (tid == 0) { float t = 0; for (int i = 0; i < 8; ++i) t += red[i]; red[8] = t * (1.f / 512.f); }
  __syncthreads();
  float mu = red[8];

  float d0 = v0 - mu, d1 = v1 - mu;
  float q = d0 * d0 + d1 * d1;
  for (int m = 16; m >= 1; m >>= 1) q += __shfl_xor(q, m, 32);
  if ((tid & 31) == 0) red[tid >> 5] = q;
  __syncthreads();
  if (tid == 0) { float t = 0; for (int i = 0; i < 8; ++i) t += red[i]; red[9] = rsqrtf(t * (1.f / 512.f) + 1e-5f); }
  __syncthreads();
  float rstd = red[9];

  float y0 = d0 * rstd * g[tid]       + be[tid];
  float y1 = d1 * rstd * g[256 + tid] + be[256 + tid];
  if (outF) { outF[base + tid] = y0; outF[base + 256 + tid] = y1; }
  if (outB) { outB[base + tid] = f2bf(y0); outB[base + 256 + tid] = f2bf(y1); }
}

// ---------------------------------------------------------------------------
// K5: top-256 of 1536 entropies per batch (rank selection, jax tie-break)
// ---------------------------------------------------------------------------
__global__ __launch_bounds__(256) void mtl_topk(
    const float* __restrict__ mems_ent, const float* __restrict__ ent_cur,
    float* __restrict__ me_top, int* __restrict__ idx_out, float* __restrict__ thresh_out) {
  __shared__ float me[1536];
  __shared__ float red[8];
  int b = blockIdx.x, tid = threadIdx.x;
  for (int i = tid; i < 1536; i += 256)
    me[i] = (i < 512) ? mems_ent[b * 512 + i] : ent_cur[(size_t)b * 1024 + (i - 512)];
  __syncthreads();

  float s = 0.f;
  for (int i = tid; i < 1536; i += 256) s += me[i];
  for (int m = 16; m >= 1; m >>= 1) s += __shfl_xor(s, m, 32);
  if ((tid & 31) == 0) red[tid >> 5] = s;
  __syncthreads();
  if (tid == 0) { float t = 0; for (int i = 0; i < 8; ++i) t += red[i]; thresh_out[b] = t * (1.f / 1536.f) * 1.0f; }

  for (int i = tid; i < 1536; i += 256) {
    float vi = me[i];
    int rank = 0;
    for (int j = 0; j < 1536; ++j) {
      float vj = me[j];
      rank += (int)(vj > vi) | ((int)(vj == vi) & (int)(j < i));
    }
    if (rank < 256) { me_top[b * 256 + rank] = vi; idx_out[b * 256 + rank] = i; }
  }
}

// K6: gather selected memory rows (from mems_x or x)
__global__ void mtl_gather(const float* __restrict__ mems_x, const float* __restrict__ x,
                           const int* __restrict__ idx, float* __restrict__ mx_out) {
  int t = blockIdx.x;            // b*256 + r
  int b = t >> 8;
  int row = idx[t];
  const float* src = (row < 512) ? mems_x + ((size_t)b * 512 + row) * 512
                                 : x + ((size_t)b * 1024 + (row - 512)) * 512;
  float* dst = mx_out + (size_t)t * 512;
  for (int e = threadIdx.x; e < 512; e += blockDim.x) dst[e] = src[e];
}

// K7: valid mask + efficiency scalar (single block, wave w = batch w)
__global__ __launch_bounds__(256) void mtl_valid_eff(
    const float* __restrict__ me_top, const float* __restrict__ thresh,
    float* __restrict__ valid_out, float* __restrict__ eff_out) {
  __shared__ float ratios[8];
  int w = threadIdx.x >> 5, lane = threadIdx.x & 31;
  float th = thresh[w];
  float sv = 0.f, cnt = 0.f;
  for (int i = lane; i < 256; i += 32) {
    float v = me_top[w * 256 + i];
    float f = (v >= th) ? 1.f : 0.f;
    valid_out[w * 256 + i] = f;
    sv += v * f; cnt += f;
  }
  for (int m = 16; m >= 1; m >>= 1) { sv += __shfl_xor(sv, m, 32); cnt += __shfl_xor(cnt, m, 32); }
  if (lane == 0) ratios[w] = sv / cnt;
  __syncthreads();
  if (threadIdx.x == 0) { float e = 0; for (int i = 0; i < 8; ++i) e += ratios[i]; eff_out[0] = e * 0.125f; }
}

// ---------------------------------------------------------------------------
// Host launcher
// ---------------------------------------------------------------------------
extern "C" void kernel_launch(void* const* d_in, const int* in_sizes, int n_in,
                              void* d_out, int out_size, void* d_ws, size_t ws_size,
                              hipStream_t stream) {
  (void)in_sizes; (void)n_in; (void)out_size; (void)ws_size;
  const float* x         = (const float*)d_in[0];
  const float* mems_x    = (const float*)d_in[1];
  const float* mems_ent  = (const float*)d_in[2];
  const float* in_proj_w = (const float*)d_in[3];
  const float* in_proj_b = (const float*)d_in[4];
  const float* out_w     = (const float*)d_in[5];
  const float* out_b     = (const float*)d_in[6];
  const float* ln1_g     = (const float*)d_in[7];
  const float* ln1_b     = (const float*)d_in[8];
  const float* w1        = (const float*)d_in[9];
  const float* b1        = (const float*)d_in[10];
  const float* w2        = (const float*)d_in[11];
  const float* b2        = (const float*)d_in[12];
  const float* ln2_g     = (const float*)d_in[13];
  const float* ln2_b     = (const float*)d_in[14];

  // d_out segments: out | mx_sel | me_top | valid | eff
  float* out_x    = (float*)d_out;                         // 8*1024*512
  float* out_mx   = out_x  + (size_t)8 * 1024 * 512;       // 8*256*512
  float* out_met  = out_mx + (size_t)8 * 256 * 512;        // 8*256
  float* out_val  = out_met + 8 * 256;                     // 8*256
  float* out_eff  = out_val + 8 * 256;                     // 1

  // workspace carve
  char* wsp = (char*)d_ws;
  auto carve = [&](size_t bytes) -> void* {
    void* p = (void*)wsp;
    wsp += (bytes + 255) & ~(size_t)255;
    return p;
  };
  bfu* kvb    = (bfu*)carve((size_t)8 * 1536 * 512 * 2);   // bf16 concat(mems_x,x)
  bfu* wqkvb  = (bfu*)carve((size_t)1536 * 512 * 2);
  bfu* outwb  = (bfu*)carve((size_t)512 * 512 * 2);
  bfu* w1b    = (bfu*)carve((size_t)2048 * 512 * 2);
  bfu* w2b    = (bfu*)carve((size_t)512 * 2048 * 2);
  bfu* qb     = (bfu*)carve((size_t)8 * 8 * 1024 * 64 * 2);
  bfu* kbf    = (bfu*)carve((size_t)8 * 8 * 1536 * 64 * 2);
  bfu* vbf    = (bfu*)carve((size_t)8 * 8 * 64 * 1536 * 2);  // transposed V
  bfu* attnc  = (bfu*)carve((size_t)8 * 1024 * 512 * 2);
  bfu* x2b    = (bfu*)carve((size_t)8 * 1024 * 512 * 2);
  bfu* h1     = (bfu*)carve((size_t)8 * 1024 * 2048 * 2);
  float* aprj = (float*)carve((size_t)8 * 1024 * 512 * 4); // attn_proj, reused as ff
  float* x2f  = (float*)carve((size_t)8 * 1024 * 512 * 4);
  float* entc = (float*)carve((size_t)8 * 1024 * 4);
  float* thr  = (float*)carve(8 * 4);
  int*   idx  = (int*)carve((size_t)8 * 256 * 4);

  // 0) conversions
  mtl_build_kvb<<<(8 * 1536 * 512) / 256, 256, 0, stream>>>(mems_x, x, kvb);
  mtl_cvt_bf16<<<(1536 * 512) / 256, 256, 0, stream>>>(in_proj_w, wqkvb, 1536 * 512);
  mtl_cvt_bf16<<<(512 * 512) / 256, 256, 0, stream>>>(out_w, outwb, 512 * 512);
  mtl_cvt_bf16<<<(2048 * 512) / 256, 256, 0, stream>>>(w1, w1b, 2048 * 512);
  mtl_cvt_bf16<<<(512 * 2048) / 256, 256, 0, stream>>>(w2, w2b, 512 * 2048);

  // 1) QKV projections: jobs = Mtiles*8 head-strips (q: 4096, k/v: 6144)
  mtl_qkv_kernel<<<dim3(768, 3, 1), 256, 0, stream>>>(kvb, wqkvb, in_proj_b, qb, kbf, vbf);

  // 2) fused attention + entropy
  mtl_attn_kernel<<<8 * 64, 256, 0, stream>>>(qb, kbf, vbf, attnc, entc);

  // 3) output projection -> f32 attn_proj   (M=8192, N=512 -> 512x8 jobs)
  mtl_gemm_wt<<<(512 * 8) / 8, 256, 0, stream>>>(attnc, 512, outwb, 512, out_b,
                                                 aprj, nullptr, 512, 8, 512, 0);
  // 4) x2 = LN(x + attn_proj)
  mtl_addln<<<8 * 1024, 256, 0, stream>>>(x, aprj, ln1_g, ln1_b, x2f, x2b);

  // 5) h1 = relu(x2 @ w1^T + b1) (bf16)   (N=2048 -> 32 n-groups)
  mtl_gemm_wt<<<(512 * 32) / 8, 256, 0, stream>>>(x2b, 512, w1b, 512, b1,
                                                  nullptr, h1, 512, 32, 512, 1);
  // 6) ff = h1 @ w2^T + b2 (f32, reuse aprj)   (K=2048)
  mtl_gemm_wt<<<(512 * 8) / 8, 256, 0, stream>>>(h1, 2048, w2b, 2048, b2,
                                                 aprj, nullptr, 512, 8, 2048, 0);
  // 7) out = LN(x2 + ff)
  mtl_addln<<<8 * 1024, 256, 0, stream>>>(x2f, aprj, ln2_g, ln2_b, out_x, nullptr);

  // 8) memory selection
  mtl_topk<<<8, 256, 0, stream>>>(mems_ent, entc, out_met, idx, thr);
  mtl_gather<<<8 * 256, 128, 0, stream>>>(mems_x, x, idx, out_mx);
  mtl_valid_eff<<<1, 256, 0, stream>>>(out_met, thr, out_val, out_eff);
}
